// HeadConvAttention_2302102470962
// MI455X (gfx1250) — matrix operand order repaired and checked by reading, not compiled, not run on hardware
//
#include <hip/hip_runtime.h>

// ---------------------------------------------------------------------------
// HeadConvAttention for gfx1250 (MI455X).
// Head-mixing folded into Q: Qp[o,q,(h,d)] = w[o,h]*scale*log2e*Q[b,h,q,d],
// giving standard flash attention with feature dim F = H*D = 1024, done with
// v_wmma_f32_16x16x32_bf16. 32-row q-tile per wave: two A-tiles share every
// K/V B-operand load. Softmax reductions via DPP16 (no LDS bpermute), exp via
// raw v_exp_f32.
// ---------------------------------------------------------------------------

typedef __attribute__((ext_vector_type(16))) __bf16 bf16x16;
typedef __attribute__((ext_vector_type(8)))  __bf16 bf16x8;
typedef __attribute__((ext_vector_type(4)))  __bf16 bf16x4;
typedef __attribute__((ext_vector_type(8)))  float  f32x8;

constexpr int B = 2, H = 16, S = 2048, D = 64;
constexpr int F = H * D;                 // 1024 folded feature dim
constexpr int QT = 32;                   // q rows per wave (2 WMMA A-tiles)
constexpr float SCALE_LOG2E = 0.125f * 1.44269504088896340736f; // 1/sqrt(64)*log2(e)

static __device__ inline bf16x16 cat16(bf16x8 lo, bf16x8 hi) {
  return __builtin_shufflevector(lo, hi, 0,1,2,3,4,5,6,7,8,9,10,11,12,13,14,15);
}

static __device__ inline f32x8 wmma_bf16(bf16x16 a, bf16x16 b, f32x8 c) {
  return __builtin_amdgcn_wmma_f32_16x16x32_bf16(false, a, false, b, (short)0, c,
                                                 false, false);
}

// raw v_exp_f32: args are always <= 0 here; underflow->0 is desired
static __device__ inline float fexp2(float x) {
#if __has_builtin(__builtin_amdgcn_exp2f)
  return __builtin_amdgcn_exp2f(x);
#else
  return exp2f(x);
#endif
}

// xor-lane reductions over a 16-lane half via DPP16 (VALU, no LDS pipe)
#if __has_builtin(__builtin_amdgcn_mov_dpp)
template <int CTRL>
static __device__ inline float dppf(float x) {
  int r = __builtin_amdgcn_mov_dpp(__builtin_bit_cast(int, x), CTRL, 0xF, 0xF, true);
  return __builtin_bit_cast(float, r);
}
static __device__ inline float red_max16(float x) {
  x = fmaxf(x, dppf<0xB1>(x));    // quad_perm [1,0,3,2]  : xor 1
  x = fmaxf(x, dppf<0x4E>(x));    // quad_perm [2,3,0,1]  : xor 2
  x = fmaxf(x, dppf<0x141>(x));   // row_half_mirror      : xor 4
  x = fmaxf(x, dppf<0x140>(x));   // row_mirror           : xor 8
  return x;
}
static __device__ inline float red_sum16(float x) {
  x += dppf<0xB1>(x);
  x += dppf<0x4E>(x);
  x += dppf<0x141>(x);
  x += dppf<0x140>(x);
  return x;
}
#else
static __device__ inline float red_max16(float x) {
  x = fmaxf(x, __shfl_xor(x, 1)); x = fmaxf(x, __shfl_xor(x, 2));
  x = fmaxf(x, __shfl_xor(x, 4)); x = fmaxf(x, __shfl_xor(x, 8));
  return x;
}
static __device__ inline float red_sum16(float x) {
  x += __shfl_xor(x, 1); x += __shfl_xor(x, 2);
  x += __shfl_xor(x, 4); x += __shfl_xor(x, 8);
  return x;
}
#endif

// ---- pack K: [B,H,S,D] f32 -> Krb [B,S, h*64+d] bf16 ----------------------
__global__ void hca_pack_k(const float* __restrict__ K, __bf16* __restrict__ Krb) {
  int i = blockIdx.x * blockDim.x + threadIdx.x;
  if (i >= B * S * F) return;
  int f = i & (F - 1);
  int s = (i / F) & (S - 1);
  int b = i / (F * S);
  int h = f >> 6, d = f & 63;
  Krb[i] = (__bf16)K[(((size_t)b * H + h) * S + s) * D + d];
}

// ---- pack V: [B,H,S,D] f32 -> Vt [B,H,D,S] bf16 (transposed for B-operand) -
__global__ void hca_pack_v(const float* __restrict__ V, __bf16* __restrict__ Vt) {
  int i = blockIdx.x * blockDim.x + threadIdx.x;
  if (i >= B * H * D * S) return;
  int s = i & (S - 1);
  int d = (i / S) & (D - 1);
  int h = (i / (S * D)) & (H - 1);
  int b = i / (S * D * H);
  Vt[i] = (__bf16)V[(((size_t)b * H + h) * S + s) * D + d];
}

// ---- flash attention, one wave32 per (b, o, 32-row q-tile) -----------------
__global__ __launch_bounds__(32) void hca_attn(const float* __restrict__ Q,
                                               const float* __restrict__ W,
                                               const __bf16* __restrict__ Krb,
                                               const __bf16* __restrict__ Vt,
                                               float* __restrict__ Out) {
  __shared__ alignas(16) __bf16 qp[QT * F];     // 64 KB: Qp tile, row-major [q][f]
  __shared__ alignas(16) __bf16 pb[QT * 32];    // 2 KB: P tile for C->A relayout

  const int lane = threadIdx.x;           // wave32: one wave per block
  const int ln   = lane & 15;
  const int khi  = lane >> 4;
  const int qt   = blockIdx.x;            // 32-row q tile
  const int o    = blockIdx.y;            // output head
  const int b    = blockIdx.z;
  const int qbase = qt * QT;

  float wloc[16];
#pragma unroll
  for (int h = 0; h < 16; ++h) wloc[h] = W[o * 16 + h] * SCALE_LOG2E;

  // Build Qp[q][h*64+d] = wloc[h] * Q[b,h,qbase+q,d] in bf16 (coalesced f32x4)
  for (int t = lane; t < (QT * F) / 4; t += 32) {
    int e = t * 4;
    int r = e >> 10;              // q row 0..31
    int f = e & (F - 1);
    int h = f >> 6, d = f & 63;
    const float4 qv = *(const float4*)(Q + (((size_t)b * H + h) * S + qbase + r) * D + d);
    float sc = wloc[h];
    bf16x4 pk = { (__bf16)(qv.x * sc), (__bf16)(qv.y * sc),
                  (__bf16)(qv.z * sc), (__bf16)(qv.w * sc) };
    *(bf16x4*)(qp + e) = pk;
  }
  // single wave: DS ops are in-order within a wave, no barrier needed

  f32x8 Oacc[2][4] = {};                  // [q-subtile][d-tile], C layout
  float mrow[2][8], lrow[2][8];           // this lane's rows: qt2*16 + khi*8 + r
#pragma unroll
  for (int t2 = 0; t2 < 2; ++t2)
#pragma unroll
    for (int r = 0; r < 8; ++r) { mrow[t2][r] = -3.0e38f; lrow[t2][r] = 0.0f; }

  const int nkb = qt + 1;                 // causal: k tiles of 32 columns
  const __bf16* kb_base = Krb + (size_t)b * S * F;
  const __bf16* v_base  = Vt + ((size_t)(b * H + o)) * D * S;
  const __bf16* arow0   = qp + ln * F + khi * 8;          // A rows 0..15
  const __bf16* arow1   = qp + (16 + ln) * F + khi * 8;   // A rows 16..31

  for (int kb = 0; kb < nkb; ++kb) {
    const int kbase = kb * 32;
    const __bf16* krow0 = kb_base + (size_t)(kbase + ln) * F + khi * 8;
    const __bf16* krow1 = krow0 + 16 * F;

    f32x8 acc[2][2] = {};                 // [q-subtile][k col-tile]
#pragma unroll 4
    for (int f0 = 0; f0 < F; f0 += 32) {
      bf16x16 a0 = cat16(*(const bf16x8*)(arow0 + f0),
                         *(const bf16x8*)(arow0 + f0 + 16));
      bf16x16 a1 = cat16(*(const bf16x8*)(arow1 + f0),
                         *(const bf16x8*)(arow1 + f0 + 16));
      bf16x16 b0 = cat16(*(const bf16x8*)(krow0 + f0),
                         *(const bf16x8*)(krow0 + f0 + 16));
      bf16x16 b1 = cat16(*(const bf16x8*)(krow1 + f0),
                         *(const bf16x8*)(krow1 + f0 + 16));
      acc[0][0] = wmma_bf16(a0, b0, acc[0][0]);   // 4 independent chains,
      acc[0][1] = wmma_bf16(a0, b1, acc[0][1]);   // every B load used twice
      acc[1][0] = wmma_bf16(a1, b0, acc[1][0]);
      acc[1][1] = wmma_bf16(a1, b1, acc[1][1]);
    }

    if (kb == nkb - 1) {                  // causal mask: diagonal tile only
#pragma unroll
      for (int t2 = 0; t2 < 2; ++t2)
#pragma unroll
        for (int r = 0; r < 8; ++r) {
          int qg = qbase + t2 * 16 + khi * 8 + r;
          if (kbase + ln > qg)      acc[t2][0][r] = -3.0e38f;
          if (kbase + 16 + ln > qg) acc[t2][1][r] = -3.0e38f;
        }
    }

    // streaming softmax (logits already in log2 domain)
#pragma unroll
    for (int t2 = 0; t2 < 2; ++t2) {
#pragma unroll
      for (int r = 0; r < 8; ++r) {
        float mx = red_max16(fmaxf(acc[t2][0][r], acc[t2][1][r]));
        float mnew = fmaxf(mrow[t2][r], mx);
        float corr = fexp2(mrow[t2][r] - mnew);
        mrow[t2][r] = mnew;
        float p0 = fexp2(acc[t2][0][r] - mnew);
        float p1 = fexp2(acc[t2][1][r] - mnew);
        acc[t2][0][r] = p0; acc[t2][1][r] = p1;
        float rs = red_sum16(p0 + p1);
        lrow[t2][r] = lrow[t2][r] * corr + rs;
#pragma unroll
        for (int dt = 0; dt < 4; ++dt) Oacc[t2][dt][r] *= corr;
      }
    }

    // P: C layout -> LDS row-major [q][32] bf16, reload in A layout
#pragma unroll
    for (int t2 = 0; t2 < 2; ++t2)
#pragma unroll
      for (int r = 0; r < 8; ++r) {
        pb[(t2 * 16 + khi * 8 + r) * 32 + ln]      = (__bf16)acc[t2][0][r];
        pb[(t2 * 16 + khi * 8 + r) * 32 + 16 + ln] = (__bf16)acc[t2][1][r];
      }
    const __bf16* prow0 = pb + ln * 32 + khi * 8;
    const __bf16* prow1 = pb + (16 + ln) * 32 + khi * 8;
    bf16x16 pa0 = cat16(*(const bf16x8*)(prow0), *(const bf16x8*)(prow0 + 16));
    bf16x16 pa1 = cat16(*(const bf16x8*)(prow1), *(const bf16x8*)(prow1 + 16));

    // PV: O[16q x 16d] += P[16x32] * V[32x16]; each V load used twice
#pragma unroll
    for (int dt = 0; dt < 4; ++dt) {
      const __bf16* vrow = v_base + (size_t)(dt * 16 + ln) * S + kbase + khi * 8;
      bf16x16 bv = cat16(*(const bf16x8*)(vrow), *(const bf16x8*)(vrow + 16));
      Oacc[0][dt] = wmma_bf16(pa0, bv, Oacc[0][dt]);
      Oacc[1][dt] = wmma_bf16(pa1, bv, Oacc[1][dt]);
    }
  }

  // epilogue: divide by row sums, store [B,H,S,D] f32
  float* orow = Out + (((size_t)(b * H + o)) * S + qbase) * D;
#pragma unroll
  for (int t2 = 0; t2 < 2; ++t2)
#pragma unroll
    for (int dt = 0; dt < 4; ++dt)
#pragma unroll
      for (int r = 0; r < 8; ++r) {
        orow[(t2 * 16 + khi * 8 + r) * D + dt * 16 + ln] =
            Oacc[t2][dt][r] / lrow[t2][r];
      }
}

extern "C" void kernel_launch(void* const* d_in, const int* in_sizes, int n_in,
                              void* d_out, int out_size, void* d_ws, size_t ws_size,
                              hipStream_t stream) {
  (void)in_sizes; (void)n_in; (void)out_size;
  const float* q = (const float*)d_in[0];
  const float* k = (const float*)d_in[1];
  const float* v = (const float*)d_in[2];
  const float* w = (const float*)d_in[3];

  const size_t krb_bytes = (size_t)B * S * F * sizeof(__bf16);     // 8 MB
  const size_t vt_bytes  = (size_t)B * H * D * S * sizeof(__bf16); // 8 MB
  if (ws_size < krb_bytes + vt_bytes) return;

  __bf16* krb = reinterpret_cast<__bf16*>(d_ws);
  __bf16* vt  = reinterpret_cast<__bf16*>((char*)d_ws + krb_bytes);

  const int totK = B * S * F;
  hca_pack_k<<<(totK + 255) / 256, 256, 0, stream>>>(k, krb);
  const int totV = B * H * D * S;
  hca_pack_v<<<(totV + 255) / 256, 256, 0, stream>>>(v, vt);

  hca_attn<<<dim3(S / QT, H, B), 32, 0, stream>>>(q, w, krb, vt, (float*)d_out);
}
